// MultiHeadedAttention_76450417869329
// MI455X (gfx1250) — compile-verified, matrix operand order checked by reading, not run on hardware
//
#include <hip/hip_runtime.h>
#include <hip/hip_bf16.h>

// ---------------------------------------------------------------------------
// MHA forward for MI455X (gfx1250): bf16 WMMA (16x16x32, f32 accum) everywhere.
//   S=4096, E=1024, H=16, Dh=64
// Pipeline (all on `stream`):
//   1) cvt_kernel x3 : fp32 -> bf16 for x, W_qkv, W_out (float4 vectorized)
//   2) qkv_gemm      : QKV = x @ W_qkv^T + b   (bf16 out, + extra V^T copy)
//   3) attn_kernel   : flash-style online-softmax attention per (head, 16-row q tile)
//   4) out_proj      : out = attn @ W_out^T + b_out (fp32 out)
// All fragment loads in a K-step are issued before any WMMA so the backend can
// stagger s_wait_loadcnt; V-fragment loads are hoisted above the softmax VALU
// block so their latency hides under exp/shfl work.
// ---------------------------------------------------------------------------

#define S_LEN 4096
#define E_DIM 1024
#define H_NUM 16
#define DH    64
#define NQKV  3072   // 3*E

typedef __attribute__((ext_vector_type(16))) __bf16 v16bf;
typedef __attribute__((ext_vector_type(8)))  __bf16 v8bf;
typedef __attribute__((ext_vector_type(4)))  __bf16 v4bf;
typedef __attribute__((ext_vector_type(8)))  float  v8f;

__device__ __forceinline__ __bf16 to_bf16(float f) {
  // round-to-nearest-even truncation fp32 -> bf16
  unsigned u = __builtin_bit_cast(unsigned, f);
  u += 0x7FFFu + ((u >> 16) & 1u);
  unsigned short hs = (unsigned short)(u >> 16);
  return __builtin_bit_cast(__bf16, hs);
}

__device__ __forceinline__ v8f wmma_bf16(v16bf a, v16bf b, v8f c) {
  // (neg_a, A, neg_b, B, c_mod, C, reuse_a, reuse_b)
  return __builtin_amdgcn_wmma_f32_16x16x32_bf16(false, a, false, b, (short)0, c,
                                                 false, false);
}

// Load a 16x32 (A) / 32x16-as-rows (B) bf16 fragment.
// Lane (hf = lane>>4, l16 = lane&15) holds 16 values:
//   element i -> row/col = l16, k = (i/8)*16 + hf*8 + (i%8)
// i.e. two contiguous 8-element (16-byte) runs per lane.
__device__ __forceinline__ v16bf load_frag(const __bf16* base, int stride,
                                           int l16, int hf, int k0) {
  union { v16bf v; v8bf h[2]; } u;
  const __bf16* p = base + (size_t)l16 * stride + k0 + hf * 8;
  u.h[0] = *(const v8bf*)(p);
  u.h[1] = *(const v8bf*)(p + 16);
  return u.v;
}

// ---------------------------------------------------------------------------
// fp32 -> bf16 conversion, 4 elements/thread (all sizes are multiples of 4)
// ---------------------------------------------------------------------------
__global__ __launch_bounds__(256) void cvt_kernel(const float4* __restrict__ src,
                                                  v4bf* __restrict__ dst, int n4) {
  int i = blockIdx.x * blockDim.x + threadIdx.x;
  int stride = gridDim.x * blockDim.x;
  for (; i < n4; i += stride) {
    float4 f = src[i];
    v4bf o;
    o[0] = to_bf16(f.x); o[1] = to_bf16(f.y);
    o[2] = to_bf16(f.z); o[3] = to_bf16(f.w);
    dst[i] = o;
  }
}

// ---------------------------------------------------------------------------
// QKV GEMM: C[S, 3E] = Xbf[S,E] @ Wbf[3E,E]^T + b
//   per-wave tile: 32 (M) x 64 (N); K-step 32; 8 wmma per step
//   Writes row-major bf16 QKV, plus V^T [H][DH][S] (16B packed stores)
// ---------------------------------------------------------------------------
__global__ __launch_bounds__(256) void qkv_gemm(const __bf16* __restrict__ xbf,
                                                const __bf16* __restrict__ wbf,
                                                const float* __restrict__ bias,
                                                __bf16* __restrict__ qkv,
                                                __bf16* __restrict__ vt) {
  const int wave = threadIdx.x >> 5;
  const int lane = threadIdx.x & 31;
  const int hf   = lane >> 4;
  const int l16  = lane & 15;
  const int wid  = blockIdx.x * 8 + wave;      // 6144 waves total
  const int m0   = (wid & 127) * 32;           // 128 M-tiles of 32
  const int n0   = (wid >> 7) * 64;            // 48  N-tiles of 64

  v8f acc[2][4] = {};
  const __bf16* a0base = xbf + (size_t)m0 * E_DIM;
  const __bf16* a1base = xbf + (size_t)(m0 + 16) * E_DIM;

  for (int k0 = 0; k0 < E_DIM; k0 += 32) {
    if (k0 + 32 < E_DIM) {
      __builtin_prefetch(a0base + (size_t)l16 * E_DIM + k0 + 32, 0, 3);
      __builtin_prefetch(a1base + (size_t)l16 * E_DIM + k0 + 32, 0, 3);
    }
    // issue ALL loads of this K-step before any WMMA (staggered waits)
    v16bf a0 = load_frag(a0base, E_DIM, l16, hf, k0);
    v16bf a1 = load_frag(a1base, E_DIM, l16, hf, k0);
    v16bf b[4];
#pragma unroll
    for (int ns = 0; ns < 4; ++ns)
      b[ns] = load_frag(wbf + (size_t)(n0 + ns * 16) * E_DIM, E_DIM, l16, hf, k0);
#pragma unroll
    for (int ns = 0; ns < 4; ++ns) {
      acc[0][ns] = wmma_bf16(a0, b[ns], acc[0][ns]);
      acc[1][ns] = wmma_bf16(a1, b[ns], acc[1][ns]);
    }
  }

#pragma unroll
  for (int ns = 0; ns < 4; ++ns) {
    const int col = n0 + ns * 16 + l16;
    const float bv = bias[col];
#pragma unroll
    for (int ms = 0; ms < 2; ++ms) {
      // row-major store (Q,K,V all)
#pragma unroll
      for (int r = 0; r < 8; ++r) {
        const int row = m0 + ms * 16 + hf * 8 + r;
        qkv[(size_t)row * NQKV + col] = to_bf16(acc[ms][ns][r] + bv);
      }
      // V columns also go to V^T [H][DH][S]; lane's 8 rows are consecutive -> 16B store
      if (col >= 2 * E_DIM) {
        const int vcol = col - 2 * E_DIM;        // head*64 + dh
        v8bf pack;
#pragma unroll
        for (int r = 0; r < 8; ++r) pack[r] = to_bf16(acc[ms][ns][r] + bv);
        const int row0 = m0 + ms * 16 + hf * 8;
        *(v8bf*)(vt + (size_t)vcol * S_LEN + row0) = pack;
      }
    }
  }
}

// ---------------------------------------------------------------------------
// Flash attention: one wave = (head, 16-query tile). Q frags resident.
//   per 32-key chunk: 4 wmma for scores (Dh=64 -> two K-steps x two key tiles),
//   online softmax with shfl_xor row reductions, P staged C->A via LDS,
//   4 wmma for P@V using V^T. V loads hoisted above softmax to hide latency.
// ---------------------------------------------------------------------------
__global__ __launch_bounds__(256) void attn_kernel(const __bf16* __restrict__ qkv,
                                                   const __bf16* __restrict__ vt,
                                                   __bf16* __restrict__ attnbf) {
  __shared__ __align__(16) __bf16 pbuf[8][16 * 32];   // 1KB per wave
  const int wave = threadIdx.x >> 5;
  const int lane = threadIdx.x & 31;
  const int hf   = lane >> 4;
  const int l16  = lane & 15;
  const int wid  = blockIdx.x * 8 + wave;             // 4096 waves
  const int head = wid >> 8;
  const int q0   = (wid & 255) << 4;

  const __bf16* qbase = qkv + (size_t)q0 * NQKV + head * DH;
  const v16bf aQ0 = load_frag(qbase, NQKV, l16, hf, 0);
  const v16bf aQ1 = load_frag(qbase, NQKV, l16, hf, 32);

  v8f oacc[4] = {};
  float mr[8], lr[8];
#pragma unroll
  for (int r = 0; r < 8; ++r) { mr[r] = -3.0e38f; lr[r] = 0.0f; }

  __bf16* myp = pbuf[wave];
  const __bf16* vhead = vt + (size_t)head * DH * S_LEN;

  for (int key0 = 0; key0 < S_LEN; key0 += 32) {
    const __bf16* kbase = qkv + (size_t)key0 * NQKV + E_DIM + head * DH;

    // issue all 4 K-matrix fragments, then 4 score WMMAs
    v16bf bk[2][2];
#pragma unroll
    for (int kt = 0; kt < 2; ++kt) {
      bk[kt][0] = load_frag(kbase + (size_t)kt * 16 * NQKV, NQKV, l16, hf, 0);
      bk[kt][1] = load_frag(kbase + (size_t)kt * 16 * NQKV, NQKV, l16, hf, 32);
    }
    v8f sc[2] = {};
#pragma unroll
    for (int kt = 0; kt < 2; ++kt) {
      sc[kt] = wmma_bf16(aQ0, bk[kt][0], sc[kt]);
      sc[kt] = wmma_bf16(aQ1, bk[kt][1], sc[kt]);
    }

    // hoist V fragment loads: independent of scores, latency hides under softmax
    v16bf bv[4];
#pragma unroll
    for (int ns = 0; ns < 4; ++ns)
      bv[ns] = load_frag(vhead + (size_t)ns * 16 * S_LEN + key0, S_LEN, l16, hf, 0);

    // online softmax update (rows r+8*hf live across the 16 lanes of this half)
#pragma unroll
    for (int r = 0; r < 8; ++r) {
      float cmax = fmaxf(sc[0][r], sc[1][r]);
#pragma unroll
      for (int off = 8; off >= 1; off >>= 1)
        cmax = fmaxf(cmax, __shfl_xor(cmax, off, 32));
      const float mnew  = fmaxf(mr[r], cmax);
      const float scale = __expf(mr[r] - mnew);
      const float p0 = __expf(sc[0][r] - mnew);
      const float p1 = __expf(sc[1][r] - mnew);
      float rs = p0 + p1;
#pragma unroll
      for (int off = 8; off >= 1; off >>= 1)
        rs += __shfl_xor(rs, off, 32);
      lr[r] = lr[r] * scale + rs;
      mr[r] = mnew;
#pragma unroll
      for (int ns = 0; ns < 4; ++ns) oacc[ns][r] *= scale;
      const int row = r + 8 * hf;
      myp[row * 32 + l16]      = to_bf16(p0);
      myp[row * 32 + 16 + l16] = to_bf16(p1);
    }

    // per-wave LDS ops are in-order; fence keeps compiler honest too
    asm volatile("s_wait_dscnt 0" ::: "memory");
    const v16bf pf = load_frag(myp, 32, l16, hf, 0);  // 16x32 A fragment

#pragma unroll
    for (int ns = 0; ns < 4; ++ns)
      oacc[ns] = wmma_bf16(pf, bv[ns], oacc[ns]);
  }

#pragma unroll
  for (int ns = 0; ns < 4; ++ns) {
    const int col = head * DH + ns * 16 + l16;
#pragma unroll
    for (int r = 0; r < 8; ++r) {
      const int row = q0 + 8 * hf + r;
      attnbf[(size_t)row * E_DIM + col] = to_bf16(oacc[ns][r] / lr[r]);
    }
  }
}

// ---------------------------------------------------------------------------
// Output projection: out[S,E] = attn_bf @ Wout_bf^T + b_out  (fp32 out)
// ---------------------------------------------------------------------------
__global__ __launch_bounds__(256) void out_proj(const __bf16* __restrict__ abf,
                                                const __bf16* __restrict__ wbf,
                                                const float* __restrict__ bias,
                                                float* __restrict__ out) {
  const int wave = threadIdx.x >> 5;
  const int lane = threadIdx.x & 31;
  const int hf   = lane >> 4;
  const int l16  = lane & 15;
  const int wid  = blockIdx.x * 8 + wave;      // 2048 waves
  const int m0   = (wid & 127) * 32;
  const int n0   = (wid >> 7) * 64;            // 16 N-tiles

  v8f acc[2][4] = {};
  const __bf16* a0base = abf + (size_t)m0 * E_DIM;
  const __bf16* a1base = abf + (size_t)(m0 + 16) * E_DIM;

  for (int k0 = 0; k0 < E_DIM; k0 += 32) {
    if (k0 + 32 < E_DIM) {
      __builtin_prefetch(a0base + (size_t)l16 * E_DIM + k0 + 32, 0, 3);
    }
    v16bf a0 = load_frag(a0base, E_DIM, l16, hf, k0);
    v16bf a1 = load_frag(a1base, E_DIM, l16, hf, k0);
    v16bf b[4];
#pragma unroll
    for (int ns = 0; ns < 4; ++ns)
      b[ns] = load_frag(wbf + (size_t)(n0 + ns * 16) * E_DIM, E_DIM, l16, hf, k0);
#pragma unroll
    for (int ns = 0; ns < 4; ++ns) {
      acc[0][ns] = wmma_bf16(a0, b[ns], acc[0][ns]);
      acc[1][ns] = wmma_bf16(a1, b[ns], acc[1][ns]);
    }
  }

#pragma unroll
  for (int ns = 0; ns < 4; ++ns) {
    const int col = n0 + ns * 16 + l16;
    const float bv = bias[col];
#pragma unroll
    for (int ms = 0; ms < 2; ++ms) {
#pragma unroll
      for (int r = 0; r < 8; ++r) {
        const int row = m0 + ms * 16 + hf * 8 + r;
        out[(size_t)row * E_DIM + col] = acc[ms][ns][r] + bv;
      }
    }
  }
}

// ---------------------------------------------------------------------------
extern "C" void kernel_launch(void* const* d_in, const int* in_sizes, int n_in,
                              void* d_out, int out_size, void* d_ws, size_t ws_size,
                              hipStream_t stream) {
  (void)in_sizes; (void)n_in; (void)out_size; (void)ws_size;
  const float* x     = (const float*)d_in[0];
  const float* Wqkv  = (const float*)d_in[1];
  const float* bqkv  = (const float*)d_in[2];
  const float* Wout  = (const float*)d_in[3];
  const float* bout  = (const float*)d_in[4];
  float* out = (float*)d_out;

  char* ws = (char*)d_ws;
  // workspace layout (bytes), all 256-aligned
  __bf16* xbf    = (__bf16*)(ws + 0);           //  8,388,608  x bf16
  __bf16* wqkvbf = (__bf16*)(ws + 8388608);     //  6,291,456  W_qkv bf16
  __bf16* woutbf = (__bf16*)(ws + 14680064);    //  2,097,152  W_out bf16
  __bf16* qkvbf  = (__bf16*)(ws + 16777216);    // 25,165,824  QKV bf16 [S,3E]
  __bf16* vtbf   = (__bf16*)(ws + 41943040);    //  8,388,608  V^T [H][DH][S]
  __bf16* attnbf = (__bf16*)(ws + 50331648);    //  8,388,608  attn out bf16 [S,E]
  // total 58,720,256 bytes

  cvt_kernel<<<1024, 256, 0, stream>>>((const float4*)x,    (v4bf*)xbf,    S_LEN * E_DIM / 4);
  cvt_kernel<<<1024, 256, 0, stream>>>((const float4*)Wqkv, (v4bf*)wqkvbf, NQKV * E_DIM / 4);
  cvt_kernel<<<512,  256, 0, stream>>>((const float4*)Wout, (v4bf*)woutbf, E_DIM * E_DIM / 4);

  // 6144 waves / 8 per block
  qkv_gemm<<<768, 256, 0, stream>>>(xbf, wqkvbf, bqkv, qkvbf, vtbf);
  // 16 heads * 256 q-tiles = 4096 waves / 8
  attn_kernel<<<512, 256, 0, stream>>>(qkvbf, vtbf, attnbf);
  // 2048 waves / 8
  out_proj<<<256, 256, 0, stream>>>(attnbf, woutbf, bout, out);
}
